// MktyTimeSeriesPredictor_70952859730395
// MI455X (gfx1250) — matrix operand order repaired and checked by reading, not compiled, not run on hardware
//
#include <hip/hip_runtime.h>
#include <hip/hip_bf16.h>

// ---------------------------------------------------------------------------
// MI455X (gfx1250) implementation.
//  - LSTM (the FLOP-dominant part) runs as a persistent per-batch-tile kernel:
//    16 rows per workgroup, h/c state resident across all 80 steps, all GEMMs
//    via v_wmma_f32_16x16x32_bf16 (bf16 A from LDS, bf16 pre-transposed
//    weights from L2-resident workspace, f32 accumulate).
//  - Text projection (8192x1024 @ 1024x256 -> 256x128) also via WMMA.
//  - Small per-row DFT / attention / gating / iDFT on VALU (trivial FLOPs).
// ---------------------------------------------------------------------------

typedef __bf16 bf16_t;
typedef __attribute__((ext_vector_type(16))) __bf16 bf16x16;
typedef __attribute__((ext_vector_type(8)))  float  f32x8;

struct __align__(16) U4 { unsigned x, y, z, w; };
struct __align__(32) F8 { U4 lo, hi; };

// A fragment (16x32 bf16, M x K): lane m=l%16, half=l/16.
// VGPR0..3 = K[half*8 .. half*8+7], VGPR4..7 = K[16+half*8 .. +7].
// p must point at row m, element (k0 + half*8) of a row-major bf16 matrix.
__device__ __forceinline__ bf16x16 frag_a(const bf16_t* p) {
  F8 f;
  f.lo = *reinterpret_cast<const U4*>(p);
  f.hi = *reinterpret_cast<const U4*>(p + 16);
  return __builtin_bit_cast(bf16x16, f);
}

// B fragment (32x16 bf16, K x N): lane holds column n = n0 + l%16,
// elements 0..7 = K[half*8 .. +7], 8..15 = K[16+half*8 .. +7]. p must point at
// element ((k0 + half*8)*ldn + n0 + l%16) of a row-major (K x N) bf16 matrix.
__device__ __forceinline__ bf16x16 frag_b(const bf16_t* p, int ldn) {
  bf16x16 v;
#pragma unroll
  for (int s = 0; s < 8; ++s) {
    v[s] = p[s * ldn];
    v[s + 8] = p[(s + 16) * ldn];
  }
  return v;
}

__device__ __forceinline__ f32x8 wmma_bf16(bf16x16 a, bf16x16 b, f32x8 c) {
  return __builtin_amdgcn_wmma_f32_16x16x32_bf16(false, a, false, b, (short)0, c,
                                                 false, false);
}

__device__ __forceinline__ f32x8 zero8() {
  f32x8 z = {0.f, 0.f, 0.f, 0.f, 0.f, 0.f, 0.f, 0.f};
  return z;
}
__device__ __forceinline__ float sigf(float x) { return 1.0f / (1.0f + __expf(-x)); }
__device__ __forceinline__ float lrelu(float x) { return x > 0.0f ? x : 0.01f * x; }

#define FPI 3.14159265358979323846f

// ---------------------------------------------------------------------------
// Prep kernels: transpose f32 (N x K) weight -> bf16 (K x N); bias fusion.
// ---------------------------------------------------------------------------
__global__ void k_transpose_bf16(const float* __restrict__ in, bf16_t* __restrict__ out,
                                 int N, int K) {
  int idx = blockIdx.x * blockDim.x + threadIdx.x;
  if (idx < N * K) {
    int n = idx / K, k = idx - n * K;
    out[k * N + n] = (bf16_t)in[idx];
  }
}

__global__ void k_add_bias(const float* __restrict__ a, const float* __restrict__ b,
                           float* __restrict__ o, int n) {
  int i = blockIdx.x * blockDim.x + threadIdx.x;
  if (i < n) o[i] = a[i] + b[i];
}

// ---------------------------------------------------------------------------
// Persistent LSTM kernel: 16 batch rows / workgroup, 8 waves, 80 steps.
// Wave w owns gate columns [16w, 16w+16) of each 128-wide gate section.
// ---------------------------------------------------------------------------
__global__ __launch_bounds__(256, 2) void lstm_kernel(
    const float* __restrict__ data,      // (8192, 64)
    const float* __restrict__ w_ih0,     // (512)
    const float* __restrict__ b0,        // (512) b_ih0 + b_hh0
    const bf16_t* __restrict__ whh0_t,   // (128, 512) bf16
    const bf16_t* __restrict__ wih1_t,   // (128, 512) bf16
    const bf16_t* __restrict__ whh1_t,   // (128, 512) bf16
    const float* __restrict__ b1,        // (512) b_ih1 + b_hh1
    const bf16_t* __restrict__ fc1_t,    // (128, 128) bf16
    const float* __restrict__ fc1_b,     // (128)
    const float* __restrict__ fc2_w,     // (128)
    const float* __restrict__ fc2_b,     // (1)
    float* __restrict__ res)             // (8192, 16)
{
  __shared__ bf16_t h0s[16][128];
  __shared__ bf16_t h1s[16][128];
  __shared__ float  hmf[16][128];
  __shared__ float  xs[16];

  const int tid  = threadIdx.x;
  const int lane = tid & 31;
  const int wave = tid >> 5;
  const int half = lane >> 4;
  const int l16  = lane & 15;
  const int ncol = (wave << 4) + l16;    // 0..127, column within a gate section
  const int row0 = blockIdx.x * 16;

  // keep the per-step weight stream warm in L2
  __builtin_prefetch(whh0_t, 0, 0);
  __builtin_prefetch(wih1_t, 0, 0);
  __builtin_prefetch(whh1_t, 0, 0);

  for (int i = tid; i < 16 * 128; i += 256) {
    (&h0s[0][0])[i] = (bf16_t)0.0f;
    (&h1s[0][0])[i] = (bf16_t)0.0f;
  }

  // per-lane constants (column ncol of each gate section)
  const float wI = w_ih0[ncol], wF = w_ih0[128 + ncol],
              wG = w_ih0[256 + ncol], wO = w_ih0[384 + ncol];
  const float bI0 = b0[ncol], bF0 = b0[128 + ncol], bG0 = b0[256 + ncol], bO0 = b0[384 + ncol];
  const float bI1 = b1[ncol], bF1 = b1[128 + ncol], bG1 = b1[256 + ncol], bO1 = b1[384 + ncol];
  const float bFc = fc1_b[ncol];
  const float bF2 = fc2_b[0];

  float c0v[8], c1v[8];
#pragma unroll
  for (int r = 0; r < 8; ++r) { c0v[r] = 0.0f; c1v[r] = 0.0f; }

  __syncthreads();

  for (int t = 0; t < 64 + 16; ++t) {
    const bool decode = (t >= 64);
    if (t < 64) {
      if (tid < 16) xs[tid] = data[(row0 + tid) * 64 + t];
    } else if (t == 64) {
      if (tid < 16) xs[tid] = data[(row0 + tid) * 64 + 63];
    }
    __syncthreads();

    // ---------------- layer 0: gates0 = h0 @ whh0^T + x*w_ih0 + b0 --------
    f32x8 ai = zero8(), af = zero8(), ag = zero8(), ao = zero8();
#pragma unroll
    for (int kk = 0; kk < 4; ++kk) {
      bf16x16 a = frag_a(&h0s[l16][kk * 32 + half * 8]);
      const int krow = kk * 32 + (half << 3);
      const bf16_t* bp = whh0_t + krow * 512 + (wave << 4) + l16;
      ai = wmma_bf16(a, frag_b(bp, 512), ai);
      af = wmma_bf16(a, frag_b(bp + 128, 512), af);
      ag = wmma_bf16(a, frag_b(bp + 256, 512), ag);
      ao = wmma_bf16(a, frag_b(bp + 384, 512), ao);
    }
    float xm[8];
#pragma unroll
    for (int r = 0; r < 8; ++r) xm[r] = xs[r + (half << 3)];
    bf16_t h0n[8];
#pragma unroll
    for (int r = 0; r < 8; ++r) {
      float gi = ai[r] + xm[r] * wI + bI0;
      float gf = af[r] + xm[r] * wF + bF0;
      float gg = ag[r] + xm[r] * wG + bG0;
      float go = ao[r] + xm[r] * wO + bO0;
      float c  = sigf(gf) * c0v[r] + sigf(gi) * tanhf(gg);
      c0v[r]   = c;
      h0n[r]   = (bf16_t)(sigf(go) * tanhf(c));
    }
    __syncthreads();
#pragma unroll
    for (int r = 0; r < 8; ++r) h0s[r + (half << 3)][ncol] = h0n[r];
    __syncthreads();

    // ------------- layer 1: gates1 = h0 @ wih1^T + h1 @ whh1^T + b1 -------
    ai = zero8(); af = zero8(); ag = zero8(); ao = zero8();
#pragma unroll
    for (int kk = 0; kk < 4; ++kk) {
      bf16x16 a0 = frag_a(&h0s[l16][kk * 32 + half * 8]);
      bf16x16 a1 = frag_a(&h1s[l16][kk * 32 + half * 8]);
      const int krow = kk * 32 + (half << 3);
      const bf16_t* bp0 = wih1_t + krow * 512 + (wave << 4) + l16;
      const bf16_t* bp1 = whh1_t + krow * 512 + (wave << 4) + l16;
      ai = wmma_bf16(a0, frag_b(bp0, 512), ai);
      ai = wmma_bf16(a1, frag_b(bp1, 512), ai);
      af = wmma_bf16(a0, frag_b(bp0 + 128, 512), af);
      af = wmma_bf16(a1, frag_b(bp1 + 128, 512), af);
      ag = wmma_bf16(a0, frag_b(bp0 + 256, 512), ag);
      ag = wmma_bf16(a1, frag_b(bp1 + 256, 512), ag);
      ao = wmma_bf16(a0, frag_b(bp0 + 384, 512), ao);
      ao = wmma_bf16(a1, frag_b(bp1 + 384, 512), ao);
    }
    bf16_t h1n[8];
#pragma unroll
    for (int r = 0; r < 8; ++r) {
      float gi = ai[r] + bI1;
      float gf = af[r] + bF1;
      float gg = ag[r] + bG1;
      float go = ao[r] + bO1;
      float c  = sigf(gf) * c1v[r] + sigf(gi) * tanhf(gg);
      c1v[r]   = c;
      h1n[r]   = (bf16_t)(sigf(go) * tanhf(c));
    }
    __syncthreads();
#pragma unroll
    for (int r = 0; r < 8; ++r) h1s[r + (half << 3)][ncol] = h1n[r];
    __syncthreads();

    // ---------------- decode: fc1 (WMMA) -> leaky_relu -> fc2 dot ---------
    if (decode) {
      f32x8 am = zero8();
#pragma unroll
      for (int kk = 0; kk < 4; ++kk) {
        bf16x16 a = frag_a(&h1s[l16][kk * 32 + half * 8]);
        const int krow = kk * 32 + (half << 3);
        am = wmma_bf16(a, frag_b(fc1_t + krow * 128 + (wave << 4) + l16, 128), am);
      }
#pragma unroll
      for (int r = 0; r < 8; ++r)
        hmf[r + (half << 3)][ncol] = lrelu(am[r] + bFc);
      __syncthreads();
      if (tid < 16) {
        float s = bF2;
        for (int k = 0; k < 128; ++k) s += hmf[tid][k] * fc2_w[k];
        res[(row0 + tid) * 16 + (t - 64)] = s;
        xs[tid] = s;   // feedback: next decode input
      }
      __syncthreads();
    }
  }
}

// ---------------------------------------------------------------------------
// Fusion kernel: text projection (WMMA), DFT, attention, gating, iDFT, mix.
// 16 rows per workgroup.
// ---------------------------------------------------------------------------
__global__ __launch_bounds__(256, 1) void fusion_kernel(
    const float* __restrict__ data,
    const float* __restrict__ text,
    const bf16_t* __restrict__ tp1_t,   // (1024, 256) bf16
    const float* __restrict__ tp1_b,    // (256)
    const bf16_t* __restrict__ tp2_t,   // (256, 128) bf16
    const float* __restrict__ tp2_b,    // (128)
    const float* __restrict__ q_w, const float* __restrict__ q_b,
    const float* __restrict__ k_w, const float* __restrict__ k_b,
    const float* __restrict__ v_w, const float* __restrict__ v_b,
    const float* __restrict__ g1_w, const float* __restrict__ g1_b,
    const float* __restrict__ g2_w, const float* __restrict__ g2_b,
    const float* __restrict__ res_lstm, // (8192, 16)
    float* __restrict__ out)            // (8192, 16)
{
  __shared__ bf16_t textb[16][1024];
  __shared__ bf16_t u[16][256];
  __shared__ float  tpf[16][128];
  __shared__ float  ds[16][64];
  __shared__ float  feat[16][64];
  __shared__ float  qk[16][64];
  __shared__ float  vvs[16][64];
  __shared__ float  gt[16][128];
  __shared__ float  gate[16][16];

  const int tid  = threadIdx.x;
  const int lane = tid & 31;
  const int wave = tid >> 5;
  const int half = lane >> 4;
  const int l16  = lane & 15;
  const int ncol = (wave << 4) + l16;
  const int row0 = blockIdx.x * 16;

  // ---- stage inputs: text -> bf16 LDS, data -> f32 LDS -------------------
  for (int i = tid; i < 16 * 1024; i += 256) {
    int r = i >> 10, c = i & 1023;
    textb[r][c] = (bf16_t)text[(size_t)(row0 + r) * 1024 + c];
  }
  for (int i = tid; i < 16 * 64; i += 256) {
    int r = i >> 6, c = i & 63;
    ds[r][c] = data[(row0 + r) * 64 + c];
  }
  __syncthreads();

  // ---- tp1: (16x1024) @ (1024x256), leaky_relu, -> bf16 LDS --------------
  {
    const int n0a = wave * 32, n0b = n0a + 16;
    f32x8 acc0 = zero8(), acc1 = zero8();
    for (int kk = 0; kk < 32; ++kk) {
      bf16x16 a = frag_a(&textb[l16][kk * 32 + half * 8]);
      const int krow = kk * 32 + (half << 3);
      const bf16_t* bp = tp1_t + krow * 256 + l16;
      acc0 = wmma_bf16(a, frag_b(bp + n0a, 256), acc0);
      acc1 = wmma_bf16(a, frag_b(bp + n0b, 256), acc1);
    }
    float bb0 = tp1_b[n0a + l16], bb1 = tp1_b[n0b + l16];
#pragma unroll
    for (int r = 0; r < 8; ++r) {
      int m = r + (half << 3);
      u[m][n0a + l16] = (bf16_t)lrelu(acc0[r] + bb0);
      u[m][n0b + l16] = (bf16_t)lrelu(acc1[r] + bb1);
    }
  }
  __syncthreads();

  // ---- tp2: (16x256) @ (256x128) -> tp (f32 LDS) -------------------------
  {
    f32x8 acc = zero8();
    for (int kk = 0; kk < 8; ++kk) {
      bf16x16 a = frag_a(&u[l16][kk * 32 + half * 8]);
      const int krow = kk * 32 + (half << 3);
      acc = wmma_bf16(a, frag_b(tp2_t + krow * 128 + (wave << 4) + l16, 128), acc);
    }
    float bb = tp2_b[ncol];
#pragma unroll
    for (int r = 0; r < 8; ++r) tpf[r + (half << 3)][ncol] = acc[r] + bb;
  }
  __syncthreads();

  // ---- 64-point rDFT: feat = [amp[1..32], phase[1..32]] ------------------
  for (int i = tid; i < 16 * 32; i += 256) {
    int row = i >> 5, f = (i & 31) + 1;
    float re = 0.f, im = 0.f;
    for (int n = 0; n < 64; ++n) {
      float sn, cs;
      __sincosf(-2.0f * FPI * (float)(f * n) * (1.0f / 64.0f), &sn, &cs);
      float d = ds[row][n];
      re += d * cs;
      im += d * sn;
    }
    feat[row][f - 1]  = sqrtf(re * re + im * im);
    feat[row][31 + f] = atan2f(im, re);
  }
  __syncthreads();

  // ---- Q, K, V -----------------------------------------------------------
  for (int i = tid; i < 16 * 64; i += 256) {
    int row = i >> 6, n = i & 63;
    float q = q_b[n];
    for (int k = 0; k < 64; ++k) q += feat[row][k] * q_w[n * 64 + k];
    float kv = k_b[n], vv = v_b[n];
    for (int k = 0; k < 128; ++k) {
      float tv = tpf[row][k];
      kv += tv * k_w[n * 128 + k];
      vv += tv * v_w[n * 128 + k];
    }
    qk[row][n]  = q * kv;
    vvs[row][n] = vv;
  }
  __syncthreads();

  // ---- adj = softmax(Q*K) * V (per row, in place in qk) ------------------
  if (tid < 16) {
    int row = tid;
    float mx = -1e30f;
    for (int n = 0; n < 64; ++n) mx = fmaxf(mx, qk[row][n]);
    float s = 0.f;
    for (int n = 0; n < 64; ++n) s += __expf(qk[row][n] - mx);
    float inv = 1.0f / s;
    for (int n = 0; n < 64; ++n)
      qk[row][n] = __expf(qk[row][n] - mx) * inv * vvs[row][n];
  }
  __syncthreads();

  // ---- gate = sigmoid(leaky_relu(adj@g1^T+b) @ g2^T + b) (16 cols) -------
  for (int i = tid; i < 16 * 128; i += 256) {
    int row = i >> 7, n = i & 127;
    float s = g1_b[n];
    for (int k = 0; k < 64; ++k) s += qk[row][k] * g1_w[n * 64 + k];
    gt[row][n] = lrelu(s);
  }
  __syncthreads();
  {
    int row = tid >> 4, n = tid & 15;
    float s = g2_b[n];
    for (int k = 0; k < 128; ++k) s += gt[row][k] * g2_w[n * 128 + k];
    gate[row][n] = sigf(s);
  }
  __syncthreads();

  // ---- irfft (first 16 samples) + final mix ------------------------------
  {
    int row = tid >> 4, n = tid & 15;
    // amp2[f] = adj[f-1] (f=1..32), ph2[f] = adj[31+f] (f=1..32), amp2[0]=0
    float acc = qk[row][31] * __cosf(qk[row][63] + FPI * (float)n);
    for (int f = 1; f < 32; ++f) {
      acc += 2.0f * qk[row][f - 1] *
             __cosf(qk[row][31 + f] + 2.0f * FPI * (float)(f * n) * (1.0f / 64.0f));
    }
    float recon = acc * (1.0f / 64.0f);
    float g = gate[row][n];
    int o = (row0 + row) * 16 + n;
    out[o] = g * res_lstm[o] + (1.0f - g) * recon;
  }
}

// ---------------------------------------------------------------------------
extern "C" void kernel_launch(void* const* d_in, const int* in_sizes, int n_in,
                              void* d_out, int out_size, void* d_ws, size_t ws_size,
                              hipStream_t stream) {
  const float* data  = (const float*)d_in[0];
  const float* text  = (const float*)d_in[1];
  const float* q_w   = (const float*)d_in[2];
  const float* q_b   = (const float*)d_in[3];
  const float* k_w   = (const float*)d_in[4];
  const float* k_b   = (const float*)d_in[5];
  const float* v_w   = (const float*)d_in[6];
  const float* v_b   = (const float*)d_in[7];
  const float* tp1_w = (const float*)d_in[8];
  const float* tp1_b = (const float*)d_in[9];
  const float* tp2_w = (const float*)d_in[10];
  const float* tp2_b = (const float*)d_in[11];
  const float* g1_w  = (const float*)d_in[12];
  const float* g1_b  = (const float*)d_in[13];
  const float* g2_w  = (const float*)d_in[14];
  const float* g2_b  = (const float*)d_in[15];
  const float* w_ih0 = (const float*)d_in[16];
  const float* b_ih0 = (const float*)d_in[17];
  const float* w_hh0 = (const float*)d_in[18];
  const float* b_hh0 = (const float*)d_in[19];
  const float* w_ih1 = (const float*)d_in[20];
  const float* b_ih1 = (const float*)d_in[21];
  const float* w_hh1 = (const float*)d_in[22];
  const float* b_hh1 = (const float*)d_in[23];
  const float* fc1_w = (const float*)d_in[24];
  const float* fc1_b = (const float*)d_in[25];
  const float* fc2_w = (const float*)d_in[26];
  const float* fc2_b = (const float*)d_in[27];

  char* ws = (char*)d_ws;
  size_t off = 0;
  auto wsAlloc = [&](size_t bytes) -> char* {
    char* p = ws + off;
    off = (off + bytes + 255) & ~(size_t)255;
    return p;
  };
  bf16_t* whh0_t = (bf16_t*)wsAlloc(128 * 512 * sizeof(bf16_t));
  bf16_t* wih1_t = (bf16_t*)wsAlloc(128 * 512 * sizeof(bf16_t));
  bf16_t* whh1_t = (bf16_t*)wsAlloc(128 * 512 * sizeof(bf16_t));
  bf16_t* fc1_t  = (bf16_t*)wsAlloc(128 * 128 * sizeof(bf16_t));
  bf16_t* tp1_t  = (bf16_t*)wsAlloc(1024 * 256 * sizeof(bf16_t));
  bf16_t* tp2_t  = (bf16_t*)wsAlloc(256 * 128 * sizeof(bf16_t));
  float*  b0     = (float*)wsAlloc(512 * sizeof(float));
  float*  b1     = (float*)wsAlloc(512 * sizeof(float));
  float*  res    = (float*)wsAlloc(8192 * 16 * sizeof(float));

  // --- weight prep ---
  k_transpose_bf16<<<(512 * 128 + 255) / 256, 256, 0, stream>>>(w_hh0, whh0_t, 512, 128);
  k_transpose_bf16<<<(512 * 128 + 255) / 256, 256, 0, stream>>>(w_ih1, wih1_t, 512, 128);
  k_transpose_bf16<<<(512 * 128 + 255) / 256, 256, 0, stream>>>(w_hh1, whh1_t, 512, 128);
  k_transpose_bf16<<<(128 * 128 + 255) / 256, 256, 0, stream>>>(fc1_w, fc1_t, 128, 128);
  k_transpose_bf16<<<(256 * 1024 + 255) / 256, 256, 0, stream>>>(tp1_w, tp1_t, 256, 1024);
  k_transpose_bf16<<<(128 * 256 + 255) / 256, 256, 0, stream>>>(tp2_w, tp2_t, 128, 256);
  k_add_bias<<<2, 256, 0, stream>>>(b_ih0, b_hh0, b0, 512);
  k_add_bias<<<2, 256, 0, stream>>>(b_ih1, b_hh1, b1, 512);

  // --- main pipeline ---
  lstm_kernel<<<8192 / 16, 256, 0, stream>>>(data, w_ih0, b0, whh0_t, wih1_t, whh1_t,
                                             b1, fc1_t, fc1_b, fc2_w, fc2_b, res);
  fusion_kernel<<<8192 / 16, 256, 0, stream>>>(data, text, tp1_t, tp1_b, tp2_t, tp2_b,
                                               q_w, q_b, k_w, k_b, v_w, v_b,
                                               g1_w, g1_b, g2_w, g2_b,
                                               res, (float*)d_out);
}
